// SVDQuantLinear_87849261072829
// MI455X (gfx1250) — matrix operand order repaired and checked by reading, not compiled, not run on hardware
//
#include <hip/hip_runtime.h>

typedef __attribute__((ext_vector_type(16))) _Float16 v16h;
typedef __attribute__((ext_vector_type(8)))  _Float16 v8h;
typedef __attribute__((ext_vector_type(8)))  float    v8f;
typedef int v4i __attribute__((vector_size(16)));   // GCC-style, matches builtin

#define M_DIM   8192
#define IN_DIM  4096
#define OUT_DIM 4096
#define RANK    32
#define GROUP   64

#if __has_builtin(__builtin_amdgcn_global_load_async_to_lds_b128) && \
    __has_builtin(__builtin_amdgcn_s_wait_asynccnt)
#define HAVE_ASYNC_LDS 1
#else
#define HAVE_ASYNC_LDS 0
#endif

// 16-byte global -> LDS copy: async (direct-to-LDS, ASYNCcnt) when available,
// else through VGPRs.
__device__ __forceinline__ void cp16(const _Float16* g, _Float16* l) {
#if HAVE_ASYNC_LDS
    __builtin_amdgcn_global_load_async_to_lds_b128(
        (__attribute__((address_space(1))) v4i*)g,
        (__attribute__((address_space(3))) v4i*)l, 0, 0);
#else
    *(v8h*)l = *(const v8h*)g;
#endif
}

__device__ __forceinline__ void stage_wait() {
#if HAVE_ASYNC_LDS
    __builtin_amdgcn_s_wait_asynccnt(0);
#endif
}

// ---------------------------------------------------------------------------
// Phase 1: W_eff[n,k] = (q[n,k]-8)*scale[n,k/64] + sum_r Lu[n,r]*Ld[r,k]
// LoRA fold via one v_wmma_f32_16x16x32_f16 per 16x16 tile (K = RANK = 32).
// ---------------------------------------------------------------------------
__global__ __launch_bounds__(256)
void svdq_dequant_fold(const int*      __restrict__ qw,
                       const _Float16* __restrict__ wscale,
                       const _Float16* __restrict__ lora_down,
                       const _Float16* __restrict__ lora_up,
                       _Float16*       __restrict__ W)
{
    const int tid  = threadIdx.x;
    const int wave = tid >> 5;
    const int lane = tid & 31;
    const int lm   = lane & 15;
    const int hi   = lane >> 4;

    const int unit    = blockIdx.x * 8 + wave;   // 0..4095 strips
    const int n_tile  = unit >> 4;               // 0..255
    const int k_strip = unit & 15;               // 0..15
    const int n0  = n_tile * 16;
    const int k0s = k_strip * 256;

    // A fragment: Lu rows n0..n0+15 K(=r) = 16*hi + s (contiguous k order, same as B)
    const _Float16* lup = lora_up + (n0 + lm) * RANK;
    v16h a;
#pragma unroll
    for (int t = 0; t < 8; ++t) {
        a[t]     = lup[16 * hi + t];
        a[8 + t] = lup[16 * hi + 8 + t];
    }

#pragma unroll 1
    for (int kt = 0; kt < 16; ++kt) {
        const int kc = k0s + kt * 16 + lm;

        // B fragment: lane holds col = lm, halfs h -> K(=r) = 16*hi + h
        v16h b;
#pragma unroll
        for (int h = 0; h < 16; ++h)
            b[h] = lora_down[(16 * hi + h) * IN_DIM + kc];

        v8f c = {};
        c = __builtin_amdgcn_wmma_f32_16x16x32_f16(
                false, a, false, b, (short)0, c, false, false);

        // D VGPR v -> element (n = n0 + v + 8*hi, k = kc)
#pragma unroll
        for (int v = 0; v < 8; ++v) {
            const int n = n0 + v + 8 * hi;
            const int q = qw[n * IN_DIM + kc];
            const float s = (float)wscale[n * (IN_DIM / GROUP) + (kc >> 6)];
            W[n * IN_DIM + kc] = (_Float16)((float)(q - 8) * s + c[v]);
        }
    }
}

// ---------------------------------------------------------------------------
// Phase 2: out[M, OUT] = x[M, IN] * W_eff[OUT, IN]^T   (fp16 in, fp32 acc)
// 256 threads = 8 waves, block tile 256(M) x 128(N), wave tile 64x64, BK=32.
// Double-buffered LDS; next tile staged (async-to-LDS when available) while
// current tile feeds 16 WMMAs per wave; one barrier per K-step.
// ---------------------------------------------------------------------------
#define BM 256
#define BN 128
#define BK 32
#define KSTR (BK + 8)   // 40 halfs = 80B row stride (bank-conflict padding)

__device__ __forceinline__ void stage(
    _Float16 As[][KSTR], _Float16 Bs[][KSTR],
    const _Float16* __restrict__ x, const _Float16* __restrict__ W,
    int mBlock, int nBlock, int kb, int r4, int c8)
{
#pragma unroll
    for (int p = 0; p < 4; ++p) {
        const int row = p * 64 + r4;
        cp16(&x[(size_t)(mBlock + row) * IN_DIM + kb + c8], &As[row][c8]);
    }
#pragma unroll
    for (int p = 0; p < 2; ++p) {
        const int row = p * 64 + r4;
        cp16(&W[(size_t)(nBlock + row) * IN_DIM + kb + c8], &Bs[row][c8]);
    }
}

__global__ __launch_bounds__(256)
void svdq_gemm(const _Float16* __restrict__ x,
               const _Float16* __restrict__ W,
               _Float16*       __restrict__ out)
{
    __shared__ _Float16 As[2][BM][KSTR];
    __shared__ _Float16 Bs[2][BN][KSTR];

    const int tid  = threadIdx.x;
    const int wave = tid >> 5;
    const int lane = tid & 31;
    const int lm   = lane & 15;
    const int hi   = lane >> 4;
    const int mw   = (wave >> 1) * 64;   // wave M offset in block tile
    const int nw   = (wave & 1) * 64;    // wave N offset in block tile

    const int mBlock = blockIdx.x * BM;
    const int nBlock = blockIdx.y * BN;

    const int r4 = tid >> 2;        // 0..63 : row within a 64-row pass
    const int c8 = (tid & 3) * 8;   // 0,8,16,24 : half offset (16B chunks)

    const v8f vzero = {};
    v8f acc[4][4];
#pragma unroll
    for (int i = 0; i < 4; ++i)
#pragma unroll
        for (int j = 0; j < 4; ++j) acc[i][j] = vzero;

    // Prologue: stage first K-slab into buffer 0
    stage(As[0], Bs[0], x, W, mBlock, nBlock, 0, r4, c8);
    stage_wait();
    __syncthreads();

    int buf = 0;
#pragma unroll 1
    for (int kb = 0; kb < IN_DIM; kb += BK) {
        // Prefetch next K-slab into the other buffer (overlaps with WMMAs)
        if (kb + BK < IN_DIM)
            stage(As[buf ^ 1], Bs[buf ^ 1], x, W, mBlock, nBlock,
                  kb + BK, r4, c8);

        // ---- build fragments: row/col = lm, K = 16*hi + s (contiguous, A and B alike) ----
        v16h afr[4], bfr[4];
#pragma unroll
        for (int i = 0; i < 4; ++i) {
            const _Float16* pa = &As[buf][mw + i * 16 + lm][16 * hi];
            v8h lo = *(const v8h*)pa;          // K = 16*hi .. +7
            v8h hp = *(const v8h*)(pa + 8);    // K = 16*hi+8 .. +15
            afr[i] = __builtin_shufflevector(lo, hp,
                     0,1,2,3,4,5,6,7,8,9,10,11,12,13,14,15);
        }
#pragma unroll
        for (int j = 0; j < 4; ++j) {
            const _Float16* pb = &Bs[buf][nw + j * 16 + lm][16 * hi];
            v8h lo = *(const v8h*)pb;          // K = 16*hi .. +7
            v8h hp = *(const v8h*)(pb + 8);    // K = 16*hi+8 .. +15
            bfr[j] = __builtin_shufflevector(lo, hp,
                     0,1,2,3,4,5,6,7,8,9,10,11,12,13,14,15);
        }

        // ---- 16 WMMAs per wave per K-step ----
#pragma unroll
        for (int i = 0; i < 4; ++i)
#pragma unroll
            for (int j = 0; j < 4; ++j)
                acc[i][j] = __builtin_amdgcn_wmma_f32_16x16x32_f16(
                    false, afr[i], false, bfr[j], (short)0, acc[i][j],
                    false, false);

        // My async writes to buf^1 must land; everyone done reading buf.
        stage_wait();
        __syncthreads();
        buf ^= 1;
    }

    // ---- write back fp16: D VGPR v -> (m = tile + v + 8*hi, n = tile + lm) ----
#pragma unroll
    for (int i = 0; i < 4; ++i) {
#pragma unroll
        for (int v = 0; v < 8; ++v) {
            const int gm = mBlock + mw + i * 16 + v + 8 * hi;
            _Float16* orow = out + (size_t)gm * OUT_DIM + nBlock + nw + lm;
#pragma unroll
            for (int j = 0; j < 4; ++j)
                orow[j * 16] = (_Float16)acc[i][j][v];
        }
    }
}

// ---------------------------------------------------------------------------
extern "C" void kernel_launch(void* const* d_in, const int* in_sizes, int n_in,
                              void* d_out, int out_size, void* d_ws, size_t ws_size,
                              hipStream_t stream) {
    const _Float16* x      = (const _Float16*)d_in[0];
    const int*      qw     = (const int*)d_in[1];
    const _Float16* wscale = (const _Float16*)d_in[2];
    const _Float16* ld     = (const _Float16*)d_in[3];
    const _Float16* lu     = (const _Float16*)d_in[4];

    _Float16* Weff = (_Float16*)d_ws;          // needs OUT*IN*2 = 32 MB scratch
    _Float16* out  = (_Float16*)d_out;

    // Phase 1: 4096 strips (256 n-tiles x 16 k-strips), 8 waves per block
    svdq_dequant_fold<<<512, 256, 0, stream>>>(qw, wscale, ld, lu, Weff);

    // Phase 2: grid 32 x 32 block tiles of 256x128
    dim3 grid(M_DIM / BM, OUT_DIM / BN);
    svdq_gemm<<<grid, 256, 0, stream>>>(x, Weff, out);
}
